// MultiHeadAttention_17076789969259
// MI455X (gfx1250) — matrix operand order repaired and checked by reading, not compiled, not run on hardware
//
#include <hip/hip_runtime.h>
#include <hip/hip_bf16.h>
#include <stdint.h>

// ---------------- problem constants ----------------
constexpr int B_  = 4;
constexpr int T_  = 2048;
constexpr int D_  = 2048;
constexpr int H_  = 16;
constexpr int KV_ = 4;
constexpr int DK_ = 128;
constexpr float ROPE_BASE_ = 10000.0f;

// ---------------- WMMA types ----------------
typedef __attribute__((ext_vector_type(16))) __bf16 bf16x16;
typedef __attribute__((ext_vector_type(8)))  float  f32x8;
typedef __attribute__((ext_vector_type(4)))  int    i32x4;

union ABFrag {
    bf16x16 bf;
    uint4   q[2];   // two 16B halves
};

static __device__ __forceinline__ f32x8 wmma_bf16(const ABFrag& a, const ABFrag& b, f32x8 c) {
    return __builtin_amdgcn_wmma_f32_16x16x32_bf16(
        /*neg_a=*/false, a.bf, /*neg_b=*/false, b.bf,
        /*c_mod=*/(short)0, c, /*reuse_a=*/false, /*reuse_b=*/false);
}

// ---------------- CDNA5 async global->LDS copy (ASYNCcnt-tracked) ----------------
// Builtin signature (probe-confirmed): (v4i AS1*, v4i AS3*, Imm offset, Imm cpol)
typedef __attribute__((address_space(1))) i32x4 glob_i32x4;
typedef __attribute__((address_space(3))) i32x4 lds_i32x4;

static __device__ __forceinline__ void async_b128(const __bf16* gsrc, __bf16* ldst) {
#if __has_builtin(__builtin_amdgcn_global_load_async_to_lds_b128)
    __builtin_amdgcn_global_load_async_to_lds_b128(
        (glob_i32x4*)gsrc, (lds_i32x4*)ldst, /*offset=*/0, /*cpol=*/0);
#else
    unsigned loff = (unsigned)(unsigned long long)(lds_i32x4*)ldst;
    asm volatile("global_load_async_to_lds_b128 %0, %1, off"
                 :: "v"(loff), "v"(gsrc) : "memory");
#endif
}

static __device__ __forceinline__ void wait_async0() {
#if __has_builtin(__builtin_amdgcn_s_wait_asynccnt)
    __builtin_amdgcn_s_wait_asynccnt(0);
#else
    asm volatile("s_wait_asynccnt 0x0" ::: "memory");
#endif
}

// ---------------- cast f32 -> bf16 ----------------
__global__ void cast_bf16_kernel(const float* __restrict__ in, __bf16* __restrict__ out, size_t n) {
    size_t i = ((size_t)blockIdx.x * blockDim.x + threadIdx.x) * 4;
    if (i + 3 < n) {
        float4 v = *reinterpret_cast<const float4*>(in + i);
        out[i+0] = (__bf16)v.x;
        out[i+1] = (__bf16)v.y;
        out[i+2] = (__bf16)v.z;
        out[i+3] = (__bf16)v.w;
    }
}

// ---------------- generic WMMA GEMM: Y[M,N] = X[M,K] * W[N,K]^T ----------------
// Async double-buffered LDS pipeline. Block 256 thr (8 waves), tile 256x128, K-step 32.
// Wave grid 4x2 -> each wave 64x64 (4x4 fragments, 16 WMMAs per 16 ds_load_b128).
__global__ __launch_bounds__(256) void gemm_bf16_nt(
    const __bf16* __restrict__ X, const __bf16* __restrict__ W,
    float* __restrict__ Y, int M, int N, int K)
{
    __shared__ __bf16 As[2][256][40];   // rows padded: 80B stride, 16B aligned
    __shared__ __bf16 Bs[2][128][40];

    const int tid  = threadIdx.x;
    const int lane = tid & 31;
    const int wid  = tid >> 5;
    const int half = lane >> 4;
    const int l16  = lane & 15;
    const int wm   = wid >> 1;   // 0..3 -> 64 rows each
    const int wn   = wid & 1;    // 0..1 -> 64 cols each

    const int m0 = blockIdx.y * 256;
    const int n0 = blockIdx.x * 128;

    // staging map: A = 1024 chunks of 16B (4/thread), B = 512 chunks (2/thread)
    const int r0  = tid >> 2;          // base row (0..63), step 64
    const int cc0 = (tid & 3) * 8;     // col in elements

    f32x8 acc[4][4] = {};

    auto stage = [&](int buf, int k0) {
        #pragma unroll
        for (int it = 0; it < 4; ++it)
            async_b128(X + (size_t)(m0 + r0 + it * 64) * K + k0 + cc0,
                       &As[buf][r0 + it * 64][cc0]);
        #pragma unroll
        for (int it = 0; it < 2; ++it)
            async_b128(W + (size_t)(n0 + r0 + it * 64) * K + k0 + cc0,
                       &Bs[buf][r0 + it * 64][cc0]);
    };

    const int nk = K >> 5;
    stage(0, 0);
    wait_async0();
    __syncthreads();

    for (int ki = 0; ki < nk; ++ki) {
        const int cur = ki & 1;
        if (ki + 1 < nk) {
            stage((ki + 1) & 1, (ki + 1) << 5);            // overlap with WMMAs below
            if (ki + 2 < nk) {                             // L2 hint two tiles ahead
                __builtin_prefetch(X + (size_t)(m0 + tid) * K + ((ki + 2) << 5), 0, 3);
                __builtin_prefetch(W + (size_t)(n0 + (tid & 127)) * K + ((ki + 2) << 5), 0, 3);
            }
        }

        // A fragments: 16x32 bf16; lane<16 -> K {0..7,16..23}, lane>=16 -> K {8..15,24..31}
        ABFrag a[4], b[4];
        #pragma unroll
        for (int fm = 0; fm < 4; ++fm) {
            const __bf16* rowp = &As[cur][wm * 64 + fm * 16 + l16][0];
            a[fm].q[0] = *reinterpret_cast<const uint4*>(rowp + half * 8);
            a[fm].q[1] = *reinterpret_cast<const uint4*>(rowp + 16 + half * 8);
        }
        // B fragments: 32x16; lane<16 -> K {0..7,16..23}, lane>=16 -> K {8..15,24..31}
        #pragma unroll
        for (int fn = 0; fn < 4; ++fn) {
            const __bf16* rowp = &Bs[cur][wn * 64 + fn * 16 + l16][0];
            b[fn].q[0] = *reinterpret_cast<const uint4*>(rowp + half * 8);
            b[fn].q[1] = *reinterpret_cast<const uint4*>(rowp + 16 + half * 8);
        }
        #pragma unroll
        for (int fm = 0; fm < 4; ++fm)
            #pragma unroll
            for (int fn = 0; fn < 4; ++fn)
                acc[fm][fn] = wmma_bf16(a[fm], b[fn], acc[fm][fn]);

        wait_async0();      // next tile landed in LDS
        __syncthreads();
    }

    // C layout: VGPR i holds M = i + 8*half, N = l16
    #pragma unroll
    for (int fm = 0; fm < 4; ++fm) {
        #pragma unroll
        for (int fn = 0; fn < 4; ++fn) {
            int rbase = m0 + wm * 64 + fm * 16 + half * 8;
            int col   = n0 + wn * 64 + fn * 16 + l16;
            #pragma unroll
            for (int i = 0; i < 8; ++i)
                Y[(size_t)(rbase + i) * N + col] = acc[fm][fn][i];
        }
    }
}

// ---------------- RoPE + cast + head-transpose ----------------
// in: f32 [B, T, nh*DK]  ->  out: bf16 [B, nh, T, DK], rotated, scaled
__global__ void rope_cast_kernel(const float* __restrict__ in, __bf16* __restrict__ out,
                                 int nh, float scale)
{
    size_t idx   = (size_t)blockIdx.x * blockDim.x + threadIdx.x;
    size_t total = (size_t)B_ * T_ * nh * (DK_ / 2);
    if (idx >= total) return;
    int    d  = (int)(idx % (DK_ / 2));
    size_t r  = idx / (DK_ / 2);
    int    hh = (int)(r % nh);
    size_t bt = r / nh;                  // b*T + t
    int    t  = (int)(bt % T_);
    int    b  = (int)(bt / T_);

    const float* p = in + ((size_t)bt * nh + hh) * DK_;
    float x1 = p[2 * d], x2 = p[2 * d + 1];
    float inv = __powf(ROPE_BASE_, -(2.0f * d) / (float)DK_);
    float sn, cs;
    __sincosf((float)t * inv, &sn, &cs);
    float r1 = (x1 * cs - x2 * sn) * scale;
    float r2 = (x1 * sn + x2 * cs) * scale;

    __bf16* q = out + (((size_t)b * nh + hh) * T_ + t) * DK_;
    q[2 * d]     = (__bf16)r1;
    q[2 * d + 1] = (__bf16)r2;
}

// ---------------- V cast + transpose: f32 [B,T,KV*DK] -> bf16 [B,KV,DK,T] ----------------
// Writes 16B-vectorized along T so the flash kernel can async-stage V^T tiles directly.
__global__ void vcast_tr_kernel(const float* __restrict__ in, __bf16* __restrict__ out)
{
    size_t idx   = (size_t)blockIdx.x * blockDim.x + threadIdx.x;
    size_t total = (size_t)B_ * KV_ * DK_ * (T_ / 8);
    if (idx >= total) return;
    int    t8 = (int)(idx % (T_ / 8));
    size_t r  = idx / (T_ / 8);
    int    dk = (int)(r % DK_);
    r /= DK_;
    int    kv = (int)(r % KV_);
    int    b  = (int)(r / KV_);
    int    t0 = t8 * 8;

    union { uint4 q; __bf16 e[8]; } u;
    #pragma unroll
    for (int e = 0; e < 8; ++e)
        u.e[e] = (__bf16)in[(((size_t)b * T_ + t0 + e) * KV_ + kv) * DK_ + dk];
    *reinterpret_cast<uint4*>(out + (((size_t)b * KV_ + kv) * DK_ + dk) * T_ + t0) = u.q;
}

// ---------------- flash attention (causal, GQA) ----------------
// Qb [B,H,T,DK] (pre-scaled), Kb [B,KV,T,DK], Vt_g [B,KV,DK,T], Ob [B,T,H*DK], bf16.
// Block: 128 threads (4 waves). Q tile 64 rows (16/wave), KV tile 64.
__global__ __launch_bounds__(128) void flash_attn_kernel(
    const __bf16* __restrict__ Qb, const __bf16* __restrict__ Kb,
    const __bf16* __restrict__ Vt_g, __bf16* __restrict__ Ob)
{
    __shared__ __bf16 Ks[64][136];     // K tile (kv x dk), 272B rows
    __shared__ __bf16 Vt[128][72];     // V^T tile (dk x kv), 144B rows
    __shared__ __bf16 Ps[4][16][72];   // per-wave P scratch (16 x 64 used)

    const int tid  = threadIdx.x;
    const int lane = tid & 31;
    const int w    = tid >> 5;
    const int half = lane >> 4;
    const int l16  = lane & 15;

    const int q0  = blockIdx.x * 64;
    const int h   = blockIdx.y;
    const int b   = blockIdx.z;
    const int kvh = h / (H_ / KV_);

    // Q A-fragments for this wave's 16 rows, DK=128 -> 4 fragments of 16x32
    ABFrag qf[4];
    {
        const __bf16* qrow = Qb + ((size_t)(b * H_ + h) * T_ + q0 + w * 16 + l16) * DK_;
        #pragma unroll
        for (int kk = 0; kk < 4; ++kk) {
            qf[kk].q[0] = *reinterpret_cast<const uint4*>(qrow + kk * 32 + half * 8);
            qf[kk].q[1] = *reinterpret_cast<const uint4*>(qrow + kk * 32 + 16 + half * 8);
        }
    }

    f32x8 o[8] = {};
    float mrow[8], lrow[8];
    #pragma unroll
    for (int i = 0; i < 8; ++i) { mrow[i] = -3.0e38f; lrow[i] = 0.0f; }

    const __bf16* kbase0 = Kb   + (size_t)(b * KV_ + kvh) * T_ * DK_;   // [kv_t][dk]
    const __bf16* vbase0 = Vt_g + (size_t)(b * KV_ + kvh) * DK_ * T_;   // [dk][kv_t]

    for (int j0 = 0; j0 < q0 + 64; j0 += 64) {   // causal early exit
        // ---- async-stage K tile (64 x 128): 1024 chunks of 16B ----
        #pragma unroll
        for (int c8 = 0; c8 < 8; ++c8) {
            int c   = tid + c8 * 128;
            int r   = c >> 4;
            int col = (c & 15) * 8;
            async_b128(kbase0 + (size_t)(j0 + r) * DK_ + col, &Ks[r][col]);
        }
        // ---- async-stage V^T tile (128 x 64): 1024 chunks of 16B ----
        #pragma unroll
        for (int c8 = 0; c8 < 8; ++c8) {
            int c   = tid + c8 * 128;
            int r   = c >> 3;          // dk 0..127
            int col = (c & 7) * 8;     // kv 0..56
            async_b128(vbase0 + (size_t)r * T_ + j0 + col, &Vt[r][col]);
        }
        wait_async0();
        __syncthreads();

        // ---- S = Q K^T (16 WMMAs) ----
        f32x8 s[4] = {};
        #pragma unroll
        for (int fn = 0; fn < 4; ++fn) {
            const __bf16* krow = &Ks[fn * 16 + l16][0];
            #pragma unroll
            for (int kk = 0; kk < 4; ++kk) {
                ABFrag bb;
                bb.q[0] = *reinterpret_cast<const uint4*>(krow + kk * 32 + half * 8);
                bb.q[1] = *reinterpret_cast<const uint4*>(krow + kk * 32 + 16 + half * 8);
                s[fn] = wmma_bf16(qf[kk], bb, s[fn]);
            }
        }

        // ---- causal mask ----
        #pragma unroll
        for (int fn = 0; fn < 4; ++fn) {
            int col = j0 + fn * 16 + l16;
            #pragma unroll
            for (int i = 0; i < 8; ++i) {
                int row = q0 + w * 16 + i + half * 8;
                if (col > row) s[fn][i] = -3.0e38f;
            }
        }

        // ---- online softmax (row stats shared across the 16 lanes of each half) ----
        #pragma unroll
        for (int i = 0; i < 8; ++i) {
            float mx = mrow[i];
            #pragma unroll
            for (int fn = 0; fn < 4; ++fn) mx = fmaxf(mx, s[fn][i]);
            #pragma unroll
            for (int off = 1; off < 16; off <<= 1)
                mx = fmaxf(mx, __shfl_xor(mx, off, 16));
            float corr = __expf(mrow[i] - mx);
            float rs = 0.0f;
            #pragma unroll
            for (int fn = 0; fn < 4; ++fn) {
                float p = __expf(s[fn][i] - mx);
                s[fn][i] = p;
                rs += p;
            }
            #pragma unroll
            for (int off = 1; off < 16; off <<= 1)
                rs += __shfl_xor(rs, off, 16);
            lrow[i] = lrow[i] * corr + rs;
            mrow[i] = mx;
            #pragma unroll
            for (int f = 0; f < 8; ++f) o[f][i] *= corr;
        }

        // ---- P (C layout) -> per-wave LDS row-major 16x64 ----
        #pragma unroll
        for (int fn = 0; fn < 4; ++fn)
            #pragma unroll
            for (int i = 0; i < 8; ++i)
                Ps[w][i + half * 8][fn * 16 + l16] = (__bf16)s[fn][i];

        // ---- O += P V (16 WMMAs); same-wave LDS ops stay in order ----
        #pragma unroll
        for (int kk = 0; kk < 2; ++kk) {
            ABFrag pa;
            const __bf16* prow = &Ps[w][l16][0];
            pa.q[0] = *reinterpret_cast<const uint4*>(prow + kk * 32 + half * 8);
            pa.q[1] = *reinterpret_cast<const uint4*>(prow + kk * 32 + 16 + half * 8);
            #pragma unroll
            for (int fn = 0; fn < 8; ++fn) {
                ABFrag vv;
                const __bf16* vrow = &Vt[fn * 16 + l16][0];
                vv.q[0] = *reinterpret_cast<const uint4*>(vrow + kk * 32 + half * 8);
                vv.q[1] = *reinterpret_cast<const uint4*>(vrow + kk * 32 + 16 + half * 8);
                o[fn] = wmma_bf16(pa, vv, o[fn]);
            }
        }
        __syncthreads();
    }

    // ---- epilogue: normalize, store to [B, T, H*DK] ----
    #pragma unroll
    for (int fn = 0; fn < 8; ++fn) {
        #pragma unroll
        for (int i = 0; i < 8; ++i) {
            int row = q0 + w * 16 + i + half * 8;
            int dk  = fn * 16 + l16;
            Ob[((size_t)b * T_ + row) * (H_ * DK_) + h * DK_ + dk] = (__bf16)(o[fn][i] / lrow[i]);
        }
    }
}

// ---------------- host side ----------------
extern "C" void kernel_launch(void* const* d_in, const int* in_sizes, int n_in,
                              void* d_out, int out_size, void* d_ws, size_t ws_size,
                              hipStream_t stream)
{
    (void)in_sizes; (void)n_in; (void)out_size; (void)ws_size;

    const float* x  = (const float*)d_in[0];
    // d_in[1] = attention_mask (all true) -> causal-only masking
    const float* Wq = (const float*)d_in[2];
    const float* Wk = (const float*)d_in[3];
    const float* Wv = (const float*)d_in[4];
    const float* Wo = (const float*)d_in[5];
    float* out = (float*)d_out;

    char* ws = (char*)d_ws;
    size_t off = 0;
    auto alloc = [&](size_t bytes) -> char* {
        char* p = ws + off;
        off += (bytes + 255) & ~(size_t)255;
        return p;
    };

    const size_t BT  = (size_t)B_ * T_;
    const size_t NX  = BT * D_;
    const size_t NWQ = (size_t)D_ * D_;
    const size_t NWK = (size_t)KV_ * DK_ * D_;
    const size_t NKV = BT * KV_ * DK_;

    __bf16* xb  = (__bf16*)alloc(NX * 2);
    __bf16* wqb = (__bf16*)alloc(NWQ * 2);
    __bf16* wkb = (__bf16*)alloc(NWK * 2);
    __bf16* wvb = (__bf16*)alloc(NWK * 2);
    __bf16* wob = (__bf16*)alloc(NWQ * 2);
    float*  Qf  = (float*) alloc(NX * 4);
    float*  Kf  = (float*) alloc(NKV * 4);
    float*  Vf  = (float*) alloc(NKV * 4);
    __bf16* Qbf = (__bf16*)alloc(NX * 2);
    __bf16* Kbf = (__bf16*)alloc(NKV * 2);
    __bf16* Vbt = (__bf16*)alloc(NKV * 2);   // [B,KV,DK,T]
    __bf16* Obf = (__bf16*)alloc(NX * 2);

    // 1) bf16 casts
    cast_bf16_kernel<<<(unsigned)((NX  / 4 + 255) / 256), 256, 0, stream>>>(x,  xb,  NX);
    cast_bf16_kernel<<<(unsigned)((NWQ / 4 + 255) / 256), 256, 0, stream>>>(Wq, wqb, NWQ);
    cast_bf16_kernel<<<(unsigned)((NWK / 4 + 255) / 256), 256, 0, stream>>>(Wk, wkb, NWK);
    cast_bf16_kernel<<<(unsigned)((NWK / 4 + 255) / 256), 256, 0, stream>>>(Wv, wvb, NWK);
    cast_bf16_kernel<<<(unsigned)((NWQ / 4 + 255) / 256), 256, 0, stream>>>(Wo, wob, NWQ);

    // 2) projections: Q = x Wq^T, K = x Wk^T, V = x Wv^T  (block tile 256x128)
    {
        dim3 gq(D_ / 128, (unsigned)(BT / 256));
        gemm_bf16_nt<<<gq, 256, 0, stream>>>(xb, wqb, Qf, (int)BT, D_, D_);
        dim3 gk((KV_ * DK_) / 128, (unsigned)(BT / 256));
        gemm_bf16_nt<<<gk, 256, 0, stream>>>(xb, wkb, Kf, (int)BT, KV_ * DK_, D_);
        gemm_bf16_nt<<<gk, 256, 0, stream>>>(xb, wvb, Vf, (int)BT, KV_ * DK_, D_);
    }

    // 3) RoPE (+1/sqrt(DK) folded into Q) and head transposes
    {
        const float qscale = 0.08838834764831845f;  // 1/sqrt(128)
        size_t nqp = BT * H_ * (DK_ / 2);
        rope_cast_kernel<<<(unsigned)((nqp + 255) / 256), 256, 0, stream>>>(Qf, Qbf, H_, qscale);
        size_t nkp = BT * KV_ * (DK_ / 2);
        rope_cast_kernel<<<(unsigned)((nkp + 255) / 256), 256, 0, stream>>>(Kf, Kbf, KV_, 1.0f);
        size_t nvt = (size_t)B_ * KV_ * DK_ * (T_ / 8);
        vcast_tr_kernel<<<(unsigned)((nvt + 255) / 256), 256, 0, stream>>>(Vf, Vbt);
    }

    // 4) causal flash attention
    {
        dim3 g(T_ / 64, H_, B_);
        flash_attn_kernel<<<g, 128, 0, stream>>>(Qbf, Kbf, Vbt, Obf);
    }

    // 5) output projection: out = O Wo^T (f32 result straight to d_out)
    {
        dim3 go(D_ / 128, (unsigned)(BT / 256));
        gemm_bf16_nt<<<go, 256, 0, stream>>>(Obf, wob, out, (int)BT, D_, D_);
    }
}
